// LIFLayer_72576357368389
// MI455X (gfx1250) — matrix operand order repaired and checked by reading, not compiled, not run on hardware
//
#include <hip/hip_runtime.h>

#define T_STEPS 256
#define BATCH   64
#define NIN     2048
#define NOUT    2048
#define M_TOTAL (T_STEPS * BATCH)   // 16384

#define BETA_M 0.9512294245007140f  // exp(-1/20)
#define BETA_S 0.8187307530779818f  // exp(-1/5)
#define V_TH   1.0f
#define T_REF  2

typedef __bf16 bf16;
typedef __attribute__((ext_vector_type(16))) __bf16 v16bf;
typedef __attribute__((ext_vector_type(8)))  __bf16 v8bf;
typedef __attribute__((ext_vector_type(8)))  float  v8f;

// ---------------------------------------------------------------------------
// Kernel 1: W (K x N, f32) -> Wt (N x K, bf16), LDS-tiled transpose.
// Makes B-matrix fragment loads contiguous in K (b128 vector loads).
// ---------------------------------------------------------------------------
__global__ __launch_bounds__(256)
void wt_transpose_kernel(const float* __restrict__ W, bf16* __restrict__ Wt) {
    __shared__ bf16 tile[32][33];   // +1 pad to dodge bank conflicts
    const int k0 = blockIdx.x * 32;
    const int n0 = blockIdx.y * 32;
    const int tx = threadIdx.x;     // 0..31
    const int ty = threadIdx.y;     // 0..7
#pragma unroll
    for (int r = ty; r < 32; r += 8)
        tile[r][tx] = (bf16)W[(size_t)(k0 + r) * NOUT + (n0 + tx)];
    __syncthreads();
#pragma unroll
    for (int r = ty; r < 32; r += 8)
        Wt[(size_t)(n0 + r) * NIN + (k0 + tx)] = tile[tx][r];
}

// ---------------------------------------------------------------------------
// Kernel 2: synaptic trace filter. One thread per (b, k) column; sequential
// over T (pure IIR, no cross-thread dependence). Emits bf16 X[t*B+b][k].
// ---------------------------------------------------------------------------
__global__ __launch_bounds__(256)
void trace_kernel(const float* __restrict__ S, bf16* __restrict__ X) {
    const int idx = blockIdx.x * blockDim.x + threadIdx.x; // over B*NIN
    const size_t stride = (size_t)BATCH * NIN;
    float x = 0.0f;
    size_t off = idx;
#pragma unroll 4
    for (int t = 0; t < T_STEPS; ++t) {
        x = x * BETA_S + S[off];
        X[off] = (bf16)x;
        off += stride;
    }
}

// ---------------------------------------------------------------------------
// Kernel 3: GEMM  I[M_TOTAL][NOUT] = X[M_TOTAL][NIN] * Wt[NOUT][NIN]^T
// via v_wmma_f32_16x16x32_bf16.
//   Wave tile  : 64 (M) x 64 (N) -> 4 A frags, 4 B frags, 16 accumulators
//   Block tile : 4 waves arranged 2x2 -> 128 x 128
//   K loop     : 2048 / 32 = 64 steps
// A/B fragments are addressed from a single global base pointer per matrix
// with compile-time immediate offsets so every load is global_load_b128
// (pure LOADcnt — no flat/DScnt involvement).
// Writes f32 results directly into the i_traj output slab.
// ---------------------------------------------------------------------------
__global__ __launch_bounds__(128)
void snn_gemm_wmma_kernel(const bf16* __restrict__ X,
                          const bf16* __restrict__ Wt,
                          float* __restrict__ I) {
    const int lane    = threadIdx.x & 31;
    const int wave    = threadIdx.x >> 5;     // 0..3
    const int wm      = wave & 1;             // M sub-tile
    const int wn      = wave >> 1;            // N sub-tile
    const int m0      = blockIdx.x * 128 + wm * 64;
    const int n0      = blockIdx.y * 128 + wn * 64;
    const int lrow    = lane & 15;
    const int halfSel = lane >> 4;            // 0: lanes 0-15, 1: lanes 16-31

    v8f acc[4][4];
    const v8f vzero = {0.f, 0.f, 0.f, 0.f, 0.f, 0.f, 0.f, 0.f};
#pragma unroll
    for (int i = 0; i < 4; ++i)
#pragma unroll
        for (int j = 0; j < 4; ++j)
            acc[i][j] = vzero;

    // A fragment per-lane K base: lanes 0-15 read K {k0..k0+7, k0+16..k0+23},
    // lanes 16-31 read K {k0+8..k0+15, k0+24..k0+31} (ISA 16-bit A layout).
    const int kbA = halfSel * 8;
    // B fragment per-lane K base: lane n holds column n, same K split as A.
    const int kbB = halfSel * 8;

    // Single base pointer per matrix; fragment rows reached via immediate
    // offsets (i * 16 * NIN elements = 64 KB steps, well within IOFFSET).
    const bf16* aBase = X  + (size_t)(m0 + lrow) * NIN + kbA;
    const bf16* bBase = Wt + (size_t)(n0 + lrow) * NIN + kbB;

    for (int k0 = 0; k0 < NIN; k0 += 32) {
        // hint next K tile toward the caches (global_prefetch_b8)
        __builtin_prefetch(aBase + 32, 0, 1);

        union { v16bf v; v8bf h[2]; } ua[4], ub[4];
#pragma unroll
        for (int i = 0; i < 4; ++i) {
            const bf16* p = aBase + (size_t)i * 16 * NIN;
            ua[i].h[0] = *reinterpret_cast<const v8bf*>(p);
            ua[i].h[1] = *reinterpret_cast<const v8bf*>(p + 16);
        }
#pragma unroll
        for (int j = 0; j < 4; ++j) {
            const bf16* p = bBase + (size_t)j * 16 * NIN;
            ub[j].h[0] = *reinterpret_cast<const v8bf*>(p);
            ub[j].h[1] = *reinterpret_cast<const v8bf*>(p + 16);
        }
        aBase += 32;
        bBase += 32;

#pragma unroll
        for (int i = 0; i < 4; ++i)
#pragma unroll
            for (int j = 0; j < 4; ++j)
                acc[i][j] = __builtin_amdgcn_wmma_f32_16x16x32_bf16(
                    /*neg_a=*/false, ua[i].v,
                    /*neg_b=*/false, ub[j].v,
                    /*c_mod=*/(short)0, acc[i][j],
                    /*reuse_a=*/false, /*reuse_b=*/false);
    }

    // Store: C/D layout — VGPR r: lanes 0-15 -> M=r, lanes 16-31 -> M=8+r.
#pragma unroll
    for (int i = 0; i < 4; ++i) {
#pragma unroll
        for (int j = 0; j < 4; ++j) {
            const int col = n0 + j * 16 + lrow;
#pragma unroll
            for (int r = 0; r < 8; ++r) {
                const int row = m0 + i * 16 + halfSel * 8 + r;
                I[(size_t)row * NOUT + col] = acc[i][j][r];
            }
        }
    }
}

// ---------------------------------------------------------------------------
// Kernel 4: LIF dynamics. One thread per (b, n) neuron, sequential over T.
// Reads i_syn from the i_traj slab of d_out; writes spikes and v_traj slabs.
// ---------------------------------------------------------------------------
__global__ __launch_bounds__(256)
void lif_dynamics_kernel(const float* __restrict__ Itraj,
                         float* __restrict__ spikes,
                         float* __restrict__ vtraj) {
    const int idx = blockIdx.x * blockDim.x + threadIdx.x; // over B*NOUT
    const size_t stride = (size_t)BATCH * NOUT;
    float v  = 0.0f;
    int   rf = 0;
    size_t off = idx;
#pragma unroll 4
    for (int t = 0; t < T_STEPS; ++t) {
        const float i_syn = Itraj[off];
        v = v * BETA_M + i_syn;
        const bool refr = rf > 0;
        if (refr) v = 0.0f;
        const bool spk = (v >= V_TH) && !refr;
        if (spk) v = 0.0f;
        rf = spk ? T_REF : (refr ? rf - 1 : 0);
        spikes[off] = spk ? 1.0f : 0.0f;
        vtraj[off]  = v;
        off += stride;
    }
}

// ---------------------------------------------------------------------------
// Host-side launch
// ---------------------------------------------------------------------------
extern "C" void kernel_launch(void* const* d_in, const int* in_sizes, int n_in,
                              void* d_out, int out_size, void* d_ws, size_t ws_size,
                              hipStream_t stream) {
    const float* S = (const float*)d_in[0];   // (T, B, N_IN) f32
    const float* W = (const float*)d_in[1];   // (N_IN, N_OUT) f32

    float* out = (float*)d_out;
    const size_t TBN = (size_t)T_STEPS * BATCH * NOUT;
    float* spikes = out;            // output 0: spikes (T,B,N_OUT)
    float* vtraj  = out + TBN;      // output 1: v_traj
    float* itraj  = out + 2 * TBN;  // output 2: i_traj (GEMM writes here)

    bf16* X  = (bf16*)d_ws;                            // 64 MB
    bf16* Wt = X + (size_t)T_STEPS * BATCH * NIN;      //  8 MB

    // 1) W -> Wt (bf16, transposed)
    wt_transpose_kernel<<<dim3(NIN / 32, NOUT / 32), dim3(32, 8), 0, stream>>>(W, Wt);

    // 2) synaptic traces X[t*B+b][k] (bf16)
    {
        const int n = BATCH * NIN;
        trace_kernel<<<n / 256, 256, 0, stream>>>(S, X);
    }

    // 3) I = X @ W  (bf16 WMMA, f32 accumulate) -> i_traj slab
    snn_gemm_wmma_kernel<<<dim3(M_TOTAL / 128, NOUT / 128), 128, 0, stream>>>(X, Wt, itraj);

    // 4) LIF membrane/refractory/spike scan
    {
        const int n = BATCH * NOUT;
        lif_dynamics_kernel<<<n / 256, 256, 0, stream>>>(itraj, spikes, vtraj);
    }
}
